// ColorHistogramMatchingLoss_13013750907387
// MI455X (gfx1250) — matrix operand order repaired and checked by reading, not compiled, not run on hardware
//
#include <hip/hip_runtime.h>
#include <hip/hip_bf16.h>

typedef _Float16 v16h __attribute__((ext_vector_type(16)));
typedef _Float16 v8h  __attribute__((ext_vector_type(8)));
typedef float    v8f  __attribute__((ext_vector_type(8)));

#define D_HIST 64
#define NPIX   65536            // 256*256 pixels per (batch, channel)
#define KSPLIT 16               // K-slices per GEMM -> 48*16 = 768 blocks
#define KSLICE (NPIX / KSPLIT)  // 4096 pixels per block
#define CHUNK  64               // pixels per iteration (= 2 WMMA K-steps of 32)
#define NCHUNK (KSLICE / CHUNK) // 64 iterations

__device__ __forceinline__ float fast_rcp(float x)  { return __builtin_amdgcn_rcpf(x); }
__device__ __forceinline__ float fast_sqrt(float x) { return __builtin_amdgcn_sqrtf(x); }

// ---------------------------------------------------------------- kernel 1
__global__ void chml_zero_kernel(float* __restrict__ hist, float* __restrict__ out, int n) {
    int i = blockIdx.x * blockDim.x + threadIdx.x;
    if (i < n) hist[i] = 0.0f;
    if (i == 0) out[0] = 0.0f;
}

// ---------------------------------------------------------------- kernel 2
// One block per (gemm, kslice). gemm = img*24 + b*3 + c.
// hist layout: [img][b][c][64][64] f32.
__global__ __launch_bounds__(256)
void chml_hist_kernel(const float* __restrict__ x, const float* __restrict__ y,
                      float* __restrict__ hist) {
    const int gid = blockIdx.x / KSPLIT;   // 0..47
    const int ks  = blockIdx.x % KSPLIT;   // K-slice
    const int img = gid / 24;
    const int b   = (gid / 3) % 8;
    const int c   = gid % 3;

    const float* src = (img == 0) ? x : y;
    const float* ch  = src + (size_t)b * 3 * NPIX;

    const int tid  = threadIdx.x;
    const int lane = tid & 31;
    const int wv   = tid >> 5;             // wave 0..7

    // Branch-free channel mixing: u = cu·log2(rgb), v = cv·log2(rgb),
    // coefficients pre-scaled by ln2 (v_log_f32 computes log2).
    const float LN2 = 0.69314718055994531f;
    // c=0: u=lr, v=lr-lb ; c=1: u=lg-lr, v=lg-lb ; c=2: u=lb-lr, v=lb-lg
    const float cu0 = (c == 0 ? 1.f : -1.f) * LN2;
    const float cu1 = (c == 1 ? 1.f : 0.f)  * LN2;
    const float cu2 = (c == 2 ? 1.f : 0.f)  * LN2;
    const float cv0 = (c == 0 ? 1.f : 0.f)  * LN2;
    const float cv1 = (c == 1 ? 1.f : (c == 2 ? -1.f : 0.f)) * LN2;
    const float cv2 = (c == 2 ? 1.f : -1.f) * LN2;

    // A[bin_u][pixel] = w * rbf_u  ;  B[bin_v][pixel] = rbf_v   (64 halves / row)
    __shared__ __align__(16) _Float16 lA[D_HIST * CHUNK];
    __shared__ __align__(16) _Float16 lB[D_HIST * CHUNK];
    __shared__ float sU[CHUNK], sV[CHUNK], sW[CHUNK];

    // 4x4 grid of 16x16 output tiles; wave wv owns tiles (mu, mv0) and (mu, mv1)
    const int mu  = wv >> 1;
    const int mv0 = (wv & 1) * 2;
    const int mv1 = mv0 + 1;

    v8f acc0 = {0.f,0.f,0.f,0.f,0.f,0.f,0.f,0.f};
    v8f acc1 = acc0;

    // RBF constants (centers pre-divided by FALL_OFF)
    const float INVF = 1.0f / 0.02f;
    const float STEP = (6.0f / 63.0f) * INVF;
    const float C0   = -3.0f * INVF;

    // RBF-fill assignment: pixel p, matrix m (0=A,1=B), bin half q (32 bins each)
    const int p = tid & 63;
    const int m = (tid >> 6) & 1;
    const int q = tid >> 7;                       // 0..1
    _Float16*    dst   = m ? lB : lA;
    const float* uvsrc = m ? sV : sU;
    const float  qoff  = (float)(q * 32) * STEP;

    // fragment addressing (this kernel's contiguous k order, same for A and B): A row M=lane%16, K halves 16*lh..+15
    //                                    B col N=lane%16, K halves 16*lh..+15
    const int lh = lane >> 4;
    const int lm = lane & 15;
    const _Float16* arow  = &lA[(mu  * 16 + lm) * CHUNK];
    const _Float16* brow0 = &lB[(mv0 * 16 + lm) * CHUNK];
    const _Float16* brow1 = &lB[(mv1 * 16 + lm) * CHUNK];

    for (int chunk = 0; chunk < NCHUNK; ++chunk) {
        __syncthreads();  // prior WMMA fragment reads done before LDS overwrite
        if (tid < CHUNK) {
            const int n = ks * KSLICE + chunk * CHUNK + tid;
            const float r  = ch[0 * NPIX + n] + 1e-6f;
            const float g  = ch[1 * NPIX + n] + 1e-6f;
            const float bl = ch[2 * NPIX + n] + 1e-6f;
            const float w  = fast_sqrt(__builtin_fmaf(r, r,
                               __builtin_fmaf(g, g, bl * bl)));
            const float l2r = __builtin_amdgcn_logf(r);
            const float l2g = __builtin_amdgcn_logf(g);
            const float l2b = __builtin_amdgcn_logf(bl);
            const float u = __builtin_fmaf(cu0, l2r, __builtin_fmaf(cu1, l2g, cu2 * l2b));
            const float v = __builtin_fmaf(cv0, l2r, __builtin_fmaf(cv1, l2g, cv2 * l2b));
            sU[tid] = u; sV[tid] = v; sW[tid] = w;
        }
        __syncthreads();
        {
            const float val  = uvsrc[p];
            const float wgt  = m ? 1.0f : sW[p];
            const float base = __builtin_fmaf(val, INVF, -C0) - qoff;
            #pragma unroll
            for (int j = 0; j < 32; ++j) {
                const float d = base - (float)j * STEP;
                const float r = wgt * fast_rcp(__builtin_fmaf(d, d, 1.0f));
                dst[(q * 32 + j) * CHUNK + p] = (_Float16)r;
            }
        }
        __syncthreads();
        // two K=32 steps: fragment loads (ds_load_b128 pairs) + 2 WMMAs each
        #pragma unroll
        for (int kk = 0; kk < 2; ++kk) {
            const int kb = kk * 32;
            union { v16h v; v8h h[2]; } fa, fb0, fb1;
            fa.h[0]  = *(const v8h*)&arow[kb + 16 * lh];
            fa.h[1]  = *(const v8h*)&arow[kb + 16 * lh + 8];
            fb0.h[0] = *(const v8h*)&brow0[kb + 16 * lh];
            fb0.h[1] = *(const v8h*)&brow0[kb + 16 * lh + 8];
            fb1.h[0] = *(const v8h*)&brow1[kb + 16 * lh];
            fb1.h[1] = *(const v8h*)&brow1[kb + 16 * lh + 8];
            acc0 = __builtin_amdgcn_wmma_f32_16x16x32_f16(false, fa.v, false, fb0.v,
                                                          (short)0, acc0, false, false);
            acc1 = __builtin_amdgcn_wmma_f32_16x16x32_f16(false, fa.v, false, fb1.v,
                                                          (short)0, acc1, false, false);
        }
    }

    // Commit partial 64x64 tile. C/D layout: lane 0-15 -> N=lane, M=VGPRr (+8 for lanes 16-31)
    float* hptr = hist + (size_t)gid * (D_HIST * D_HIST);
    #pragma unroll
    for (int r = 0; r < 8; ++r) {
        const int u = mu * 16 + lh * 8 + r;
        atomicAdd(&hptr[u * 64 + mv0 * 16 + lm], acc0[r]);
        atomicAdd(&hptr[u * 64 + mv1 * 16 + lm], acc1[r]);
    }
}

// ---------------------------------------------------------------- kernel 3
// One block per batch b: normalize both hists, Hellinger distance, mean into out.
__global__ __launch_bounds__(256)
void chml_reduce_kernel(const float* __restrict__ hist, float* __restrict__ out) {
    const int b   = blockIdx.x;
    const int tid = threadIdx.x;
    const int NB  = 3 * D_HIST * D_HIST;   // 12288 per (img, batch)
    const float* hx = hist + (size_t)(b * 3) * (D_HIST * D_HIST);
    const float* hy = hist + (size_t)(24 + b * 3) * (D_HIST * D_HIST);

    __shared__ float sA[256], sB[256];
    float tx = 0.f, ty = 0.f;
    for (int i = tid; i < NB; i += 256) { tx += hx[i]; ty += hy[i]; }
    sA[tid] = tx; sB[tid] = ty;
    __syncthreads();
    for (int s = 128; s > 0; s >>= 1) {
        if (tid < s) { sA[tid] += sA[tid + s]; sB[tid] += sB[tid + s]; }
        __syncthreads();
    }
    const float itx = 1.0f / sA[0];
    const float ity = 1.0f / sB[0];
    __syncthreads();

    float h = 0.f;
    for (int i = tid; i < NB; i += 256) {
        const float d = fast_sqrt(hy[i] * ity) - fast_sqrt(hx[i] * itx);
        h = __builtin_fmaf(d, d, h);
    }
    sA[tid] = h;
    __syncthreads();
    for (int s = 128; s > 0; s >>= 1) {
        if (tid < s) sA[tid] += sA[tid + s];
        __syncthreads();
    }
    if (tid == 0)
        atomicAdd(out, fast_sqrt(sA[0]) * 0.70710678118654752f * 0.125f);
}

// ---------------------------------------------------------------- launch
extern "C" void kernel_launch(void* const* d_in, const int* in_sizes, int n_in,
                              void* d_out, int out_size, void* d_ws, size_t ws_size,
                              hipStream_t stream) {
    const float* x = (const float*)d_in[0];
    const float* y = (const float*)d_in[1];
    float* out  = (float*)d_out;
    float* hist = (float*)d_ws;           // [2][8][3][64][64] f32 = 786432 B

    const int nhist = 2 * 8 * 3 * D_HIST * D_HIST;  // 196608
    chml_zero_kernel<<<(nhist + 255) / 256, 256, 0, stream>>>(hist, out, nhist);
    chml_hist_kernel<<<48 * KSPLIT, 256, 0, stream>>>(x, y, hist);
    chml_reduce_kernel<<<8, 256, 0, stream>>>(hist, out);
}
